// BertEncoder_47528108098085
// MI455X (gfx1250) — matrix operand order repaired and checked by reading, not compiled, not run on hardware
//
#include <hip/hip_runtime.h>
#include <hip/hip_bf16.h>

typedef _Float16 half_t;
typedef __attribute__((ext_vector_type(16))) _Float16 v16h;
typedef __attribute__((ext_vector_type(8)))  float    v8f;
typedef __attribute__((ext_vector_type(4)))  int      i32x4;

#define AS1 __attribute__((address_space(1)))
#define AS3 __attribute__((address_space(3)))

#if defined(__has_builtin)
#if __has_builtin(__builtin_amdgcn_global_load_async_to_lds_b128) && \
    __has_builtin(__builtin_amdgcn_s_wait_asynccnt)
#define USE_ASYNC_LDS 1
#endif
#endif

#define NUM_GRAPHS 128
#define MAX_NODES  512
#define DMODEL     256
#define NHEAD      8
#define DFF        1024
#define ROWS       (NUM_GRAPHS * MAX_NODES)   // 65536

__device__ __forceinline__ float gelu_exact(float x) {
    return 0.5f * x * (1.0f + erff(x * 0.70710678118654752f));
}

// ---------------- small utility kernels ----------------

__global__ void k_cvt(const float* __restrict__ s, half_t* __restrict__ d, int n) {
    int i = blockIdx.x * 256 + threadIdx.x;
    if (i < n) d[i] = (half_t)s[i];
}

__global__ void k_zero4(float4* __restrict__ p, int n4) {
    int i = blockIdx.x * 256 + threadIdx.x;
    if (i < n4) p[i] = make_float4(0.f, 0.f, 0.f, 0.f);
}

// starts[g] = lower_bound(batch, g); starts[128] = N
__global__ void k_starts(const int* __restrict__ batch, int Nn, int* __restrict__ starts) {
    int g = threadIdx.x;
    if (g > NUM_GRAPHS) return;
    int lo = 0, hi = Nn;
    while (lo < hi) { int mid = (lo + hi) >> 1; if (batch[mid] < g) lo = mid + 1; else hi = mid; }
    starts[g] = lo;
}

// one block per node, 256 threads = channels; mask-token substitution fused; dual f32/f16 write
__global__ void k_scatter(const float* __restrict__ h, const int* __restrict__ batch,
                          const unsigned char* __restrict__ mask, const float* __restrict__ token,
                          const int* __restrict__ starts, float* __restrict__ X,
                          half_t* __restrict__ X16) {
    int i = blockIdx.x, c = threadIdx.x;
    int g = batch[i];
    int p = i - starts[g];
    float v = mask[i] ? token[c] : h[(size_t)i * DMODEL + c];
    size_t idx = ((size_t)g * MAX_NODES + p) * DMODEL + c;
    X[idx] = v;
    X16[idx] = (half_t)v;
}

__global__ void k_gather(const float* __restrict__ X, const int* __restrict__ batch,
                         const int* __restrict__ starts, float* __restrict__ out,
                         half_t* __restrict__ out16) {
    int i = blockIdx.x, c = threadIdx.x;
    int g = batch[i];
    int p = i - starts[g];
    float v = X[((size_t)g * MAX_NODES + p) * DMODEL + c];
    out[(size_t)i * DMODEL + c] = v;
    out16[(size_t)i * DMODEL + c] = (half_t)v;
}

// ---------------- LayerNorm (wave per row) ----------------
// RES: x = LN(x + res); GELUF: gelu after LN; EMIT16: also write f16 mirror
template <int W, bool RES, bool GELUF, bool EMIT16>
__global__ __launch_bounds__(256) void k_ln(float* __restrict__ x, const float* __restrict__ res,
                                            const float* __restrict__ g, const float* __restrict__ b,
                                            half_t* __restrict__ x16, int rows) {
    int wave = threadIdx.x >> 5, lane = threadIdx.x & 31;
    int row = blockIdx.x * 8 + wave;
    if (row >= rows) return;
    constexpr int E = W / 32;
    float v[E];
    float* xr = x + (size_t)row * W;
    const float* rr = res + (size_t)row * W;
    float s = 0.f, s2 = 0.f;
#pragma unroll
    for (int j = 0; j < E; ++j) {
        int c = lane + j * 32;
        float t = xr[c];
        if (RES) t += rr[c];
        v[j] = t; s += t; s2 += t * t;
    }
#pragma unroll
    for (int m = 16; m >= 1; m >>= 1) { s += __shfl_xor(s, m, 32); s2 += __shfl_xor(s2, m, 32); }
    float mu = s / (float)W;
    float var = s2 / (float)W - mu * mu;
    float rs = rsqrtf(var + 1e-5f);
#pragma unroll
    for (int j = 0; j < E; ++j) {
        int c = lane + j * 32;
        float t = (v[j] - mu) * rs * g[c] + b[c];
        if (GELUF) t = gelu_exact(t);
        xr[c] = t;
        if (EMIT16) x16[(size_t)row * W + c] = (half_t)t;
    }
}

// ---------------- generic WMMA GEMM (f16 A, f16 W, f32 accum) ----------------
// C[M,N] = act(A[M,K] @ W^T + bias), W row-major [N,K] f16.
// Block tile 128x64, 8 waves (4x2) each 32x32 (2x2 WMMA frags), K step 32.
// LDS staging via GLOBAL_LOAD_ASYNC_TO_LDS_B128 when available.
template <bool OUT_F16, bool GELUF, bool MASK>
__global__ __launch_bounds__(256) void k_gemm(const half_t* __restrict__ A, const half_t* __restrict__ W,
                                              const float* __restrict__ bias, void* __restrict__ Cout,
                                              const unsigned char* __restrict__ maskp,
                                              int M, int N, int K) {
    __shared__ half_t sA[128][32];
    __shared__ half_t sB[64][32];
    int tid = threadIdx.x;
    int lane = tid & 31, wave = tid >> 5;
    int wm = wave >> 1, wn = wave & 1;
    int hi = (lane >> 4) & 1, ln15 = lane & 15;
    int m0 = blockIdx.x * 128;
    int n0 = blockIdx.y * 64;

    v8f acc[2][2];
    v8f zf = {};
    acc[0][0] = zf; acc[0][1] = zf; acc[1][0] = zf; acc[1][1] = zf;

    for (int k0 = 0; k0 < K; k0 += 32) {
        // stage A tile: 128 rows x 64B = 512 x 16B chunks, 2 per thread
#pragma unroll
        for (int u = 0; u < 2; ++u) {
            int c = tid + u * 256;
            int row = c >> 2;
            const half_t* gp = A + (size_t)(m0 + row) * K + k0 + (c & 3) * 8;
            void* lp = (char*)&sA[0][0] + (size_t)c * 16;
#ifdef USE_ASYNC_LDS
            __builtin_amdgcn_global_load_async_to_lds_b128((AS1 i32x4*)(void*)gp, (AS3 i32x4*)lp, 0, 0);
#else
            *(uint4*)lp = *(const uint4*)(const void*)gp;
#endif
        }
        // stage W tile as [n][k]: 64 x 64B = 256 x 16B chunks, 1 per thread
        {
            int c = tid;
            int nl = c >> 2;
            const half_t* gp = W + (size_t)(n0 + nl) * K + k0 + (c & 3) * 8;
            void* lp = (char*)&sB[0][0] + (size_t)c * 16;
#ifdef USE_ASYNC_LDS
            __builtin_amdgcn_global_load_async_to_lds_b128((AS1 i32x4*)(void*)gp, (AS3 i32x4*)lp, 0, 0);
#else
            *(uint4*)lp = *(const uint4*)(const void*)gp;
#endif
        }
        if (k0 + 32 < K) {  // prefetch next K-step tiles (global_prefetch_b8)
            __builtin_prefetch(A + (size_t)(m0 + (tid >> 1)) * K + k0 + 32, 0, 1);
            __builtin_prefetch(W + (size_t)(n0 + (tid >> 2)) * K + k0 + 32, 0, 1);
        }
#ifdef USE_ASYNC_LDS
        __builtin_amdgcn_s_wait_asynccnt(0);
#endif
        __syncthreads();

        v16h af[2], bf[2];
#pragma unroll
        for (int mt = 0; mt < 2; ++mt) {
            int row = wm * 32 + mt * 16 + ln15;
#pragma unroll
            for (int e = 0; e < 16; ++e) {
                int kk = e + ((e >= 8) ? 8 : 0) + (hi ? 8 : 0);  // A-fragment K mapping
                af[mt][e] = sA[row][kk];
            }
        }
#pragma unroll
        for (int nt = 0; nt < 2; ++nt) {
            int col = wn * 32 + nt * 16 + ln15;
#pragma unroll
            for (int e = 0; e < 16; ++e) {
                int kk = e + ((e >= 8) ? 8 : 0) + (hi ? 8 : 0);  // B-fragment K mapping
                bf[nt][e] = sB[col][kk];
            }
        }
#pragma unroll
        for (int mt = 0; mt < 2; ++mt)
#pragma unroll
            for (int nt = 0; nt < 2; ++nt)
                acc[mt][nt] = __builtin_amdgcn_wmma_f32_16x16x32_f16(
                    false, af[mt], false, bf[nt], (short)0, acc[mt][nt], false, false);
        __syncthreads();
    }
    // epilogue: bias (+gelu) (+row mask), f16 or f32 store
#pragma unroll
    for (int mt = 0; mt < 2; ++mt) {
#pragma unroll
        for (int nt = 0; nt < 2; ++nt) {
            int n_g = n0 + wn * 32 + nt * 16 + ln15;
            float bv = bias[n_g];
#pragma unroll
            for (int r = 0; r < 8; ++r) {
                int m_g = m0 + wm * 32 + mt * 16 + r + hi * 8;
                if (m_g < M) {
                    float v = acc[mt][nt][r] + bv;
                    if (GELUF) v = gelu_exact(v);
                    if (MASK) v = maskp[m_g] ? v : 0.f;
                    if (OUT_F16) ((half_t*)Cout)[(size_t)m_g * N + n_g] = (half_t)v;
                    else         ((float*)Cout)[(size_t)m_g * N + n_g] = v;
                }
            }
        }
    }
}

// ---------------- flash attention (1 wave per (graph, head, 16-query tile)) ----------------
// QKV layout: [ROWS][768] f16, columns [0:256)=Q, [256:512)=K, [512:768)=V, head h at h*32.
__global__ __launch_bounds__(32) void k_attention(const half_t* __restrict__ qkv,
                                                  const int* __restrict__ starts,
                                                  half_t* __restrict__ ctx16) {
    int bid = blockIdx.x;
    int qt = bid & 31; bid >>= 5;
    int hh = bid & 7;  int g = bid >> 3;
    int lane = threadIdx.x;
    int hi = (lane >> 4) & 1, ln15 = lane & 15;
    int cnt = starts[g + 1] - starts[g];
    const half_t* base = qkv + (size_t)g * MAX_NODES * 768;
    const float scale = 0.1767766952966369f;  // 1/sqrt(32)

    // Q as A-fragment (16 queries x dh=32)
    v16h aq;
    {
        const half_t* qrow = base + (size_t)(qt * 16 + ln15) * 768 + hh * 32;
#pragma unroll
        for (int e = 0; e < 16; ++e) {
            int k = e + ((e >= 8) ? 8 : 0) + (hi ? 8 : 0);
            aq[e] = qrow[k];
        }
    }

    v8f zf = {};
    v8f o0 = zf, o1 = zf;     // ctx accumulators: dh 0..15 and 16..31
    float m_r[8], l_r[8];
#pragma unroll
    for (int r = 0; r < 8; ++r) { m_r[r] = -3.0e38f; l_r[r] = 0.f; }

    __shared__ half_t sP[16][32];   // probability relayout bounce (C-layout -> A-layout)

    for (int c = 0; c < MAX_NODES; c += 32) {
        // K tiles as B-fragments: columns = keys, k = dh
        v16h b0, b1;
        {
            const half_t* k0p = base + (size_t)(c + ln15) * 768 + 256 + hh * 32;
            const half_t* k1p = base + (size_t)(c + 16 + ln15) * 768 + 256 + hh * 32;
#pragma unroll
            for (int e = 0; e < 16; ++e) {
                int kk = e + ((e >= 8) ? 8 : 0) + (hi ? 8 : 0);
                b0[e] = k0p[kk];
                b1[e] = k1p[kk];
            }
        }
        v8f s0 = __builtin_amdgcn_wmma_f32_16x16x32_f16(false, aq, false, b0, (short)0, zf, false, false);
        v8f s1 = __builtin_amdgcn_wmma_f32_16x16x32_f16(false, aq, false, b1, (short)0, zf, false, false);

        bool v0 = (c + ln15) < cnt;
        bool v1 = (c + 16 + ln15) < cnt;
#pragma unroll
        for (int r = 0; r < 8; ++r) {
            float x0 = v0 ? (float)s0[r] * scale : -1e9f;
            float x1 = v1 ? (float)s1[r] * scale : -1e9f;
            float mx = fmaxf(x0, x1);
#pragma unroll
            for (int m = 8; m >= 1; m >>= 1) mx = fmaxf(mx, __shfl_xor(mx, m, 32));
            float mnew = fmaxf(m_r[r], mx);
            float alpha = __expf(m_r[r] - mnew);
            float p0 = __expf(x0 - mnew);
            float p1 = __expf(x1 - mnew);
            float rs = p0 + p1;
#pragma unroll
            for (int m = 8; m >= 1; m >>= 1) rs += __shfl_xor(rs, m, 32);
            l_r[r] = l_r[r] * alpha + rs;
            m_r[r] = mnew;
            o0[r] *= alpha;
            o1[r] *= alpha;
            int prow = r + hi * 8;
            sP[prow][ln15]      = (half_t)p0;
            sP[prow][16 + ln15] = (half_t)p1;
        }
        __syncthreads();
        // P as A-fragment (16 queries x 32 keys)
        v16h ap;
#pragma unroll
        for (int e = 0; e < 16; ++e) {
            int kk = e + ((e >= 8) ? 8 : 0) + (hi ? 8 : 0);
            ap[e] = sP[ln15][kk];
        }
        // V tiles as B-fragments: k = key-in-chunk, n = dh column
        v16h bv0, bv1;
#pragma unroll
        for (int e = 0; e < 16; ++e) {
            int key = c + e + ((e >= 8) ? 8 : 0) + (hi ? 8 : 0);
            const half_t* vrow = base + (size_t)key * 768 + 512 + hh * 32;
            bv0[e] = vrow[ln15];
            bv1[e] = vrow[16 + ln15];
        }
        o0 = __builtin_amdgcn_wmma_f32_16x16x32_f16(false, ap, false, bv0, (short)0, o0, false, false);
        o1 = __builtin_amdgcn_wmma_f32_16x16x32_f16(false, ap, false, bv1, (short)0, o1, false, false);
        __syncthreads();
    }
    // normalize and store ctx16[g, q, h*32 + n]
    half_t* crow = ctx16 + ((size_t)g * MAX_NODES + qt * 16) * DMODEL + hh * 32;
#pragma unroll
    for (int r = 0; r < 8; ++r) {
        float inv = 1.0f / l_r[r];
        int m = r + hi * 8;
        crow[(size_t)m * DMODEL + ln15]      = (half_t)(o0[r] * inv);
        crow[(size_t)m * DMODEL + 16 + ln15] = (half_t)(o1[r] * inv);
    }
}

// ---------------- host orchestration ----------------

extern "C" void kernel_launch(void* const* d_in, const int* in_sizes, int n_in,
                              void* d_out, int out_size, void* d_ws, size_t ws_size,
                              hipStream_t stream) {
    (void)n_in; (void)out_size; (void)ws_size;
    const float* h       = (const float*)d_in[0];
    const int* batch     = (const int*)d_in[1];
    const unsigned char* maskp = (const unsigned char*)d_in[2];
    const float* token   = (const float*)d_in[3];
    const float* qkv_w   = (const float*)d_in[4];
    const float* qkv_b   = (const float*)d_in[5];
    const float* out_w   = (const float*)d_in[6];
    const float* out_b   = (const float*)d_in[7];
    const float* ln1_g   = (const float*)d_in[8];
    const float* ln1_b   = (const float*)d_in[9];
    const float* ff1_w   = (const float*)d_in[10];
    const float* ff1_b   = (const float*)d_in[11];
    const float* ff2_w   = (const float*)d_in[12];
    const float* ff2_b   = (const float*)d_in[13];
    const float* ln2_g   = (const float*)d_in[14];
    const float* ln2_b   = (const float*)d_in[15];
    const float* p1_w    = (const float*)d_in[16];
    const float* p1_b    = (const float*)d_in[17];
    const float* pln1_g  = (const float*)d_in[18];
    const float* pln1_b  = (const float*)d_in[19];
    const float* p2_w    = (const float*)d_in[20];
    const float* p2_b    = (const float*)d_in[21];
    const float* pln2_g  = (const float*)d_in[22];
    const float* pln2_b  = (const float*)d_in[23];
    const float* p3_w    = (const float*)d_in[24];
    const float* p3_b    = (const float*)d_in[25];
    const int Nn = in_sizes[1];  // total nodes

    // workspace carve-out
    char* ws = (char*)d_ws;
    size_t off = 0;
    auto alloc = [&](size_t bytes) -> void* {
        void* p = ws + off;
        off += (bytes + 255) & ~(size_t)255;
        return p;
    };
    int*    starts = (int*)alloc(129 * sizeof(int));
    half_t* w_qkv  = (half_t*)alloc((size_t)2 * 768 * 256 * 2);
    half_t* w_out  = (half_t*)alloc((size_t)2 * 256 * 256 * 2);
    half_t* w_ff1  = (half_t*)alloc((size_t)2 * 1024 * 256 * 2);
    half_t* w_ff2  = (half_t*)alloc((size_t)2 * 256 * 1024 * 2);
    half_t* w_p1   = (half_t*)alloc((size_t)512 * 256 * 2);
    half_t* w_p2   = (half_t*)alloc((size_t)256 * 512 * 2);
    half_t* w_p3   = (half_t*)alloc((size_t)128 * 256 * 2);
    half_t* NE16   = (half_t*)alloc((size_t)(Nn + 128) * DMODEL * 2);  // +pad for edge-tile reads
    half_t* X16    = (half_t*)alloc((size_t)ROWS * DMODEL * 2);
    float*  X      = (float*)alloc((size_t)ROWS * DMODEL * 4);
    half_t* QKV    = (half_t*)alloc((size_t)ROWS * 768 * 2);
    half_t* CTX16  = (half_t*)alloc((size_t)ROWS * DMODEL * 2);
    float*  Y      = (float*)alloc((size_t)ROWS * DMODEL * 4);
    half_t* FF     = (half_t*)alloc((size_t)ROWS * DFF * 2);
    float*  H1     = (float*)FF;      // head hidden1 [Nn,512] f32 reuses FF region
    half_t* H116   = QKV;             // f16 mirror of H1 reuses QKV region
    float*  H2     = Y;               // head hidden2 [Nn,256] f32 reuses Y region
    half_t* H216   = CTX16;           // f16 mirror of H2 reuses CTX16 region

    auto cvt = [&](const float* s, half_t* d, int n) {
        k_cvt<<<(n + 255) / 256, 256, 0, stream>>>(s, d, n);
    };
    cvt(qkv_w, w_qkv, 2 * 768 * 256);
    cvt(out_w, w_out, 2 * 256 * 256);
    cvt(ff1_w, w_ff1, 2 * 1024 * 256);
    cvt(ff2_w, w_ff2, 2 * 256 * 1024);
    cvt(p1_w,  w_p1,  512 * 256);
    cvt(p2_w,  w_p2,  256 * 512);
    cvt(p3_w,  w_p3,  128 * 256);

    k_starts<<<1, 129, 0, stream>>>(batch, Nn, starts);
    k_zero4<<<(ROWS * DMODEL / 4 + 255) / 256, 256, 0, stream>>>((float4*)X, ROWS * DMODEL / 4);
    k_zero4<<<(ROWS * DMODEL / 8 + 255) / 256, 256, 0, stream>>>((float4*)X16, ROWS * DMODEL / 8);
    k_scatter<<<Nn, 256, 0, stream>>>(h, batch, maskp, token, starts, X, X16);

    for (int l = 0; l < 2; ++l) {
        dim3 gq(ROWS / 128, 768 / 64);
        k_gemm<true, false, false><<<gq, 256, 0, stream>>>(
            X16, w_qkv + (size_t)l * 768 * 256, qkv_b + l * 768, QKV, nullptr, ROWS, 768, 256);
        k_attention<<<NUM_GRAPHS * NHEAD * (MAX_NODES / 16), 32, 0, stream>>>(QKV, starts, CTX16);
        dim3 go(ROWS / 128, 256 / 64);
        k_gemm<false, false, false><<<go, 256, 0, stream>>>(
            CTX16, w_out + (size_t)l * 256 * 256, out_b + l * 256, Y, nullptr, ROWS, 256, 256);
        k_ln<256, true, false, true><<<ROWS / 8, 256, 0, stream>>>(
            X, Y, ln1_g + l * 256, ln1_b + l * 256, X16, ROWS);
        dim3 gf1(ROWS / 128, DFF / 64);
        k_gemm<true, true, false><<<gf1, 256, 0, stream>>>(
            X16, w_ff1 + (size_t)l * DFF * 256, ff1_b + l * DFF, FF, nullptr, ROWS, DFF, 256);
        dim3 gf2(ROWS / 128, 256 / 64);
        k_gemm<false, false, false><<<gf2, 256, 0, stream>>>(
            FF, w_ff2 + (size_t)l * 256 * DFF, ff2_b + l * 256, Y, nullptr, ROWS, 256, 1024);
        k_ln<256, true, false, true><<<ROWS / 8, 256, 0, stream>>>(
            X, Y, ln2_g + l * 256, ln2_b + l * 256, X16, ROWS);
    }

    float* node_emb = (float*)d_out;                 // [Nn, 256]
    float* outp = node_emb + (size_t)Nn * DMODEL;    // [Nn, 128]
    k_gather<<<Nn, 256, 0, stream>>>(X, batch, starts, node_emb, NE16);

    dim3 gp1((Nn + 127) / 128, 512 / 64);
    k_gemm<false, false, false><<<gp1, 256, 0, stream>>>(
        NE16, w_p1, p1_b, H1, nullptr, Nn, 512, 256);
    k_ln<512, false, true, true><<<(Nn + 7) / 8, 256, 0, stream>>>(
        H1, nullptr, pln1_g, pln1_b, H116, Nn);
    dim3 gp2((Nn + 127) / 128, 256 / 64);
    k_gemm<false, false, false><<<gp2, 256, 0, stream>>>(
        H116, w_p2, p2_b, H2, nullptr, Nn, 256, 512);
    k_ln<256, false, true, true><<<(Nn + 7) / 8, 256, 0, stream>>>(
        H2, nullptr, pln2_g, pln2_b, H216, Nn);
    dim3 gp3((Nn + 127) / 128, 128 / 64);
    k_gemm<false, false, true><<<gp3, 256, 0, stream>>>(
        H216, w_p3, p3_b, outp, maskp, Nn, 128, 256);
}
